// S4CondBlock_24275155157031
// MI455X (gfx1250) — matrix operand order repaired and checked by reading, not compiled, not run on hardware
//
#include <hip/hip_runtime.h>
#include <cstdint>
#include <cstddef>

// Problem constants (from reference): B=16, H=512, L=2048, N=64, CD=128
#define BB    16
#define HH    512
#define LL    2048
#define NMODE 64
#define CDIM  128

typedef __attribute__((ext_vector_type(8)))  float          v8f;
typedef __attribute__((ext_vector_type(16))) __bf16         v16bf;
typedef __attribute__((ext_vector_type(16))) unsigned short v16us;

union FragB16 {
  v16bf b;
  v16us u;
  uint4 q[2];
};

static __device__ __forceinline__ unsigned short f32_to_bf16_bits(float f) {
  union { float f; unsigned u; } a; a.f = f;
  unsigned r = a.u + 0x7FFFu + ((a.u >> 16) & 1u);  // round-to-nearest-even
  return (unsigned short)(r >> 16);
}
static __device__ __forceinline__ float bf16_bits_to_f32(unsigned short h) {
  union { unsigned u; float f; } a; a.u = ((unsigned)h) << 16;
  return a.f;
}
static __device__ __forceinline__ float gelu_exact(float x) {
  return 0.5f * x * (1.0f + erff(x * 0.70710678118654752f));
}

// ---------------------------------------------------------------------------
// Workspace layout (bytes); total ~70.3 MB
// ---------------------------------------------------------------------------
static constexpr size_t WS_HBF   = 0;                                   // B*H*L bf16
static constexpr size_t WS_XBF   = WS_HBF + (size_t)BB * HH * LL * 2;   // B*H*L bf16
static constexpr size_t WS_WBF   = WS_XBF + (size_t)BB * HH * LL * 2;   // H*H bf16
static constexpr size_t WS_KBF   = WS_WBF + (size_t)HH * HH * 2;        // H*L bf16
static constexpr size_t WS_DTACM = WS_KBF + (size_t)HH * LL * 2;        // H*N*4 f32
static constexpr size_t WS_GB    = WS_DTACM + (size_t)HH * NMODE * 4 * 4; // B*2H f32

// ---------------------------------------------------------------------------
// Prep: convert linear_w (fp32) -> bf16 bits
// ---------------------------------------------------------------------------
__global__ void k_prep_w(const float* __restrict__ w, unsigned short* __restrict__ w_bf) {
  int i = blockIdx.x * 256 + threadIdx.x;
  if (i < HH * HH) w_bf[i] = f32_to_bf16_bits(w[i]);
}

// ---------------------------------------------------------------------------
// Prep: convert x (fp32) -> bf16, vectorized (float4 in, uint2 out)
// ---------------------------------------------------------------------------
__global__ void k_prep_x(const float* __restrict__ x, unsigned short* __restrict__ x_bf) {
  size_t i = (size_t)blockIdx.x * 256 + threadIdx.x;     // one float4 per thread
  if (i >= (size_t)BB * HH * LL / 4) return;
  float4 f = *(const float4*)(x + i * 4);
  uint2 o;
  o.x = (unsigned)f32_to_bf16_bits(f.x) | ((unsigned)f32_to_bf16_bits(f.y) << 16);
  o.y = (unsigned)f32_to_bf16_bits(f.z) | ((unsigned)f32_to_bf16_bits(f.w) << 16);
  *(uint2*)(x_bf + i * 4) = o;
}

// ---------------------------------------------------------------------------
// Prep: per-(h,n) S4 coefficients: dtA = A*dt, Cm = C*(exp(dtA)-1)/A
// ---------------------------------------------------------------------------
__global__ void k_prep_dtacm(const float* __restrict__ log_dt,
                             const float* __restrict__ C_ri,
                             const float* __restrict__ log_A_real,
                             const float* __restrict__ A_imag,
                             float* __restrict__ dtacm) {
  int idx = blockIdx.x * 256 + threadIdx.x;
  if (idx >= HH * NMODE) return;
  int g = idx >> 6;
  float dt   = __expf(log_dt[g]);
  float a_re = -__expf(log_A_real[idx]);
  float a_im = A_imag[idx];
  float dre  = a_re * dt;
  float dim  = a_im * dt;
  float e    = __expf(dre);
  float em_re = e * cosf(dim) - 1.0f;
  float em_im = e * sinf(dim);
  float c_re = C_ri[2 * idx + 0];
  float c_im = C_ri[2 * idx + 1];
  float n_re = c_re * em_re - c_im * em_im;           // C * (exp(dtA)-1)
  float n_im = c_re * em_im + c_im * em_re;
  float inv  = 1.0f / (a_re * a_re + a_im * a_im);    // / A
  float cm_re = (n_re * a_re + n_im * a_im) * inv;
  float cm_im = (n_im * a_re - n_re * a_im) * inv;
  dtacm[4 * idx + 0] = dre;
  dtacm[4 * idx + 1] = dim;
  dtacm[4 * idx + 2] = cm_re;
  dtacm[4 * idx + 3] = cm_im;
}

// ---------------------------------------------------------------------------
// Prep: FiLM  gb[b, j] = sum_c cond[b,c]*film_w[j,c] + film_b[j]   (16 x 1024)
// ---------------------------------------------------------------------------
__global__ void k_film(const float* __restrict__ cond,
                       const float* __restrict__ film_w,
                       const float* __restrict__ film_b,
                       float* __restrict__ gb) {
  int idx = blockIdx.x * 256 + threadIdx.x;
  if (idx >= BB * 2 * HH) return;
  int b = idx >> 10;
  int j = idx & 1023;
  const float* cp = cond + (size_t)b * CDIM;
  const float* wp = film_w + (size_t)j * CDIM;
  float s = film_b[j];
  #pragma unroll 8
  for (int c = 0; c < CDIM; ++c) s += cp[c] * wp[c];
  gb[idx] = s;
}

// ---------------------------------------------------------------------------
// S4 kernel: K[g,l] = 2 * sum_n Re(Cm * exp(dtA*l)) -> bf16
// ---------------------------------------------------------------------------
__global__ __launch_bounds__(256) void k_s4kernel(const float* __restrict__ dtacm,
                                                  unsigned short* __restrict__ K_bf) {
  __shared__ float coef[NMODE * 4];
  int g  = blockIdx.x >> 3;
  int l  = ((blockIdx.x & 7) << 8) + threadIdx.x;
  if (threadIdx.x < NMODE * 4) coef[threadIdx.x] = dtacm[(size_t)g * NMODE * 4 + threadIdx.x];
  __syncthreads();
  float t = (float)l;
  float acc = 0.0f;
  #pragma unroll 4
  for (int n = 0; n < NMODE; ++n) {
    float dre = coef[4 * n + 0], dim = coef[4 * n + 1];
    float cre = coef[4 * n + 2], cim = coef[4 * n + 3];
    float er = __expf(dre * t);
    float s, c;
    sincosf(dim * t, &s, &c);
    acc += cre * er * c - cim * er * s;
  }
  K_bf[(size_t)g * LL + l] = f32_to_bf16_bits(2.0f * acc);
}

// ---------------------------------------------------------------------------
// h = gelu(W @ x + bias), bf16 in / bf16 out.
// grid = (H/16, L/128, B), block = 256 (8 waves), one 16x16 out tile per wave.
// WMMA: M = out-channel (g), N = time (l), K = in-channel.
// Dual accumulators (even/odd K-chunks) keep two independent WMMA streams in
// flight; next-next B row is prefetched (global_prefetch_b8).
// ---------------------------------------------------------------------------
__global__ __launch_bounds__(256) void k_gemm_gelu(const unsigned short* __restrict__ x_bf,
                                                   const unsigned short* __restrict__ w_bf,
                                                   const float* __restrict__ linear_b,
                                                   unsigned short* __restrict__ h_bf) {
  // W panel 16 x 512 bf16, padded row stride 520 (1040B % 256B banks -> 4-bank rotate)
  __shared__ unsigned short Wlds[16 * 520];
  const int tid  = threadIdx.x;
  const int lane = tid & 31;
  const int wave = tid >> 5;
  const int g0   = blockIdx.x * 16;
  const int l0   = blockIdx.y * 128 + wave * 16;
  const int b    = blockIdx.z;

  for (int i = tid; i < 1024; i += 256) {  // 1024 uint4 = 16 rows * 64
    int m = i >> 6, j = i & 63;
    *(uint4*)(Wlds + m * 520 + j * 8) =
        *(const uint4*)(w_bf + (size_t)(g0 + m) * HH + j * 8);
  }
  __syncthreads();

  const int mrow  = lane & 15;
  const int kbase = (lane & 16) ? 8 : 0;  // ISA 16-bit A layout: hi-half lanes own K+8
  const unsigned short* xcol = x_bf + (size_t)b * HH * LL + l0 + (lane & 15);

  v8f c0 = {}, c1 = {};
  for (int kk = 0; kk < HH / 32; kk += 2) {
    if (kk + 2 < HH / 32)
      __builtin_prefetch(xcol + (size_t)(kk + 2) * 32 * LL, 0, 0);

    // ---- chunk kk (even) ----
    {
      FragB16 a, bf;
      const unsigned short* ap = Wlds + mrow * 520 + kk * 32 + kbase;
      a.q[0] = *(const uint4*)(ap);        // K = kbase+0..7
      a.q[1] = *(const uint4*)(ap + 16);   // K = 16+kbase+0..7
      const unsigned short* bp = xcol + (size_t)kk * 32 * LL;  // col = lane&15, row = kk*32+k
      #pragma unroll
      for (int i = 0; i < 8; ++i) { bf.u[i] = bp[(size_t)(kbase + i) * LL]; bf.u[8 + i] = bp[(size_t)(16 + kbase + i) * LL]; }
      c0 = __builtin_amdgcn_wmma_f32_16x16x32_bf16(false, a.b, false, bf.b,
                                                   (short)0, c0, false, false);
    }
    // ---- chunk kk+1 (odd) ----
    {
      FragB16 a, bf;
      const unsigned short* ap = Wlds + mrow * 520 + (kk + 1) * 32 + kbase;
      a.q[0] = *(const uint4*)(ap);
      a.q[1] = *(const uint4*)(ap + 16);
      const unsigned short* bp = xcol + (size_t)(kk + 1) * 32 * LL;
      #pragma unroll
      for (int i = 0; i < 8; ++i) { bf.u[i] = bp[(size_t)(kbase + i) * LL]; bf.u[8 + i] = bp[(size_t)(16 + kbase + i) * LL]; }
      c1 = __builtin_amdgcn_wmma_f32_16x16x32_bf16(false, a.b, false, bf.b,
                                                   (short)0, c1, false, false);
    }
  }
  v8f c = c0 + c1;

  // D layout: VGPR r, lane<16 -> M=r, lane>=16 -> M=r+8; N = lane&15
  const int gofs = (lane & 16) ? 8 : 0;
  const int ln   = lane & 15;
  #pragma unroll
  for (int r = 0; r < 8; ++r) {
    int g = g0 + r + gofs;
    float v = c[r] + linear_b[g];
    h_bf[((size_t)b * HH + g) * LL + l0 + ln] = f32_to_bf16_bits(gelu_exact(v));
  }
}

// ---------------------------------------------------------------------------
// Per-channel causal Toeplitz conv (WMMA) + D*h + BatchNorm + FiLM + gelu + res
// grid = H (one block per channel), block = 256 (8 waves).
// WMMA mapping: M = batch (16), N = output-time tile (16), K = lag chunk (32).
// LDS: h slice (16x2048 bf16, padded stride 2056) + K row (32-zero causal
//      prefix + 2048) + y (16x2048 f32) + reduction slots. ~196 KB of the
//      WGP's 320 KB LDS -> whole channel is LDS-resident.
// ---------------------------------------------------------------------------
static constexpr int HSTR = 2056;                        // padded ushort stride
static constexpr size_t S_HL  = 0;                       // 16*2056*2 = 65792
static constexpr size_t S_KP  = 65792;                   // 2080 ushort = 4160
static constexpr size_t S_Y   = 69952;                   // 32768 f32 = 131072
static constexpr size_t S_RED = 201024;                  // 2 f32
static constexpr size_t SMEM2 = 201088;

__global__ __launch_bounds__(256) void k_conv_epilogue(
    const unsigned short* __restrict__ h_bf,
    const unsigned short* __restrict__ K_bf,
    const float* __restrict__ Dcoef,
    const float* __restrict__ gb,
    const float* __restrict__ res_w,
    const float* __restrict__ x,
    float* __restrict__ out) {
  extern __shared__ unsigned char smem[];
  unsigned short* hl = (unsigned short*)(smem + S_HL);
  unsigned short* KP = (unsigned short*)(smem + S_KP);
  float* ylds        = (float*)(smem + S_Y);
  float* red         = (float*)(smem + S_RED);

  const int g    = blockIdx.x;
  const int tid  = threadIdx.x;
  const int lane = tid & 31;
  const int wave = tid >> 5;

  // --- stage channel slice h[:, g, :] and causal-padded K row into LDS ---
  for (int i = tid; i < 4096; i += 256) {  // 4096 uint4 = 16 rows * 256
    int b = i >> 8, j = i & 255;
    *(uint4*)(hl + b * HSTR + j * 8) =
        *(const uint4*)(h_bf + ((size_t)b * HH + g) * LL + j * 8);
  }
  for (int j = tid; j < LL; j += 256) KP[32 + j] = K_bf[(size_t)g * LL + j];
  if (tid < 32) KP[tid] = 0;  // K[d<0] = 0 implements causality inside WMMA
  __syncthreads();

  const float Dg   = Dcoef[g];
  const int  mrow  = lane & 15;            // batch row for A
  const int  kbase = (lane & 16) ? 8 : 0;
  const int  gofs  = (lane & 16) ? 8 : 0;
  const int  ln    = lane & 15;

  // --- causal Toeplitz matmul: y[b, l] = sum_{t<=l} h[b,t] * K[l-t] ---
  for (int mt = wave; mt < LL / 16; mt += 8) {
    const int l0 = mt * 16;
    v8f c0 = {}, c1 = {};
    const int tcmax = (l0 + 15) >> 5;      // chunks with any t <= l
    int tc = 0;
    for (; tc + 1 <= tcmax; tc += 2) {
      // ---- chunk tc -> c0 ----
      {
        const int t0 = tc * 32;
        FragB16 a, bf;
        const unsigned short* ap = hl + mrow * HSTR + t0 + kbase;
        a.q[0] = *(const uint4*)(ap);
        a.q[1] = *(const uint4*)(ap + 16);
        const unsigned short* kp = KP + 32 + (l0 + ln - t0);
        #pragma unroll
        for (int i = 0; i < 16; ++i) bf.u[i] = kp[-(kbase + (i & 7) + ((i & 8) << 1))];
        c0 = __builtin_amdgcn_wmma_f32_16x16x32_bf16(false, a.b, false, bf.b,
                                                     (short)0, c0, false, false);
      }
      // ---- chunk tc+1 -> c1 ----
      {
        const int t0 = (tc + 1) * 32;
        FragB16 a, bf;
        const unsigned short* ap = hl + mrow * HSTR + t0 + kbase;
        a.q[0] = *(const uint4*)(ap);
        a.q[1] = *(const uint4*)(ap + 16);
        const unsigned short* kp = KP + 32 + (l0 + ln - t0);
        #pragma unroll
        for (int i = 0; i < 16; ++i) bf.u[i] = kp[-(kbase + (i & 7) + ((i & 8) << 1))];
        c1 = __builtin_amdgcn_wmma_f32_16x16x32_bf16(false, a.b, false, bf.b,
                                                     (short)0, c1, false, false);
      }
    }
    if (tc <= tcmax) {                      // odd tail chunk
      const int t0 = tc * 32;
      FragB16 a, bf;
      const unsigned short* ap = hl + mrow * HSTR + t0 + kbase;
      a.q[0] = *(const uint4*)(ap);
      a.q[1] = *(const uint4*)(ap + 16);
      const unsigned short* kp = KP + 32 + (l0 + ln - t0);
      #pragma unroll
      for (int i = 0; i < 16; ++i) bf.u[i] = kp[-(kbase + (i & 7) + ((i & 8) << 1))];
      c0 = __builtin_amdgcn_wmma_f32_16x16x32_bf16(false, a.b, false, bf.b,
                                                   (short)0, c0, false, false);
    }
    v8f c = c0 + c1;

    // D layout -> y tile (+ skip-term D*h), into LDS f32 buffer
    #pragma unroll
    for (int r = 0; r < 8; ++r) {
      int b = r + gofs;
      int l = l0 + ln;
      float hv = bf16_bits_to_f32(hl[b * HSTR + l]);
      ylds[b * LL + l] = c[r] + Dg * hv;
    }
  }
  __syncthreads();

  // --- BatchNorm stats over all (B, L) of this channel ---
  if (tid < 2) red[tid] = 0.0f;
  __syncthreads();
  float s = 0.0f, s2 = 0.0f;
  for (int i = tid; i < BB * LL; i += 256) {
    float v = ylds[i];
    s += v; s2 += v * v;
  }
  #pragma unroll
  for (int off = 16; off > 0; off >>= 1) {
    s  += __shfl_down(s, off, 32);
    s2 += __shfl_down(s2, off, 32);
  }
  if (lane == 0) {
    atomicAdd(&red[0], s);
    atomicAdd(&red[1], s2);
  }
  __syncthreads();
  const float inv_n = 1.0f / (float)(BB * LL);
  const float mean  = red[0] * inv_n;
  const float var   = red[1] * inv_n - mean * mean;
  const float rstd  = rsqrtf(var + 1e-5f);
  const float resv  = res_w[g];

  // --- normalize + FiLM + gelu + residual, write fp32 output ---
  for (int i = tid; i < BB * LL; i += 256) {
    int b = i >> 11;
    int l = i & (LL - 1);
    float nv = (ylds[i] - mean) * rstd;
    float ga = gb[(size_t)b * 2 * HH + g];
    float be = gb[(size_t)b * 2 * HH + HH + g];
    float v  = gelu_exact(nv * ga + be);
    size_t idx = ((size_t)b * HH + g) * LL + l;
    out[idx] = v + x[idx] * resv;
  }
}

// ---------------------------------------------------------------------------
extern "C" void kernel_launch(void* const* d_in, const int* in_sizes, int n_in,
                              void* d_out, int out_size, void* d_ws, size_t ws_size,
                              hipStream_t stream) {
  (void)in_sizes; (void)n_in; (void)out_size; (void)ws_size;
  const float* x          = (const float*)d_in[0];
  const float* cond       = (const float*)d_in[1];
  const float* linear_w   = (const float*)d_in[2];
  const float* linear_b   = (const float*)d_in[3];
  const float* log_dt     = (const float*)d_in[4];
  const float* C_ri       = (const float*)d_in[5];
  const float* log_A_real = (const float*)d_in[6];
  const float* A_imag     = (const float*)d_in[7];
  const float* Dcoef      = (const float*)d_in[8];
  const float* film_w     = (const float*)d_in[9];
  const float* film_b     = (const float*)d_in[10];
  const float* res_w      = (const float*)d_in[11];
  float* out = (float*)d_out;

  unsigned char* ws = (unsigned char*)d_ws;
  unsigned short* h_bf  = (unsigned short*)(ws + WS_HBF);
  unsigned short* x_bf  = (unsigned short*)(ws + WS_XBF);
  unsigned short* w_bf  = (unsigned short*)(ws + WS_WBF);
  unsigned short* K_bf  = (unsigned short*)(ws + WS_KBF);
  float*          dtacm = (float*)(ws + WS_DTACM);
  float*          gbuf  = (float*)(ws + WS_GB);

  k_prep_w    <<<(HH * HH + 255) / 256, 256, 0, stream>>>(linear_w, w_bf);
  k_prep_x    <<<(BB * HH * LL / 4 + 255) / 256, 256, 0, stream>>>(x, x_bf);
  k_prep_dtacm<<<(HH * NMODE + 255) / 256, 256, 0, stream>>>(log_dt, C_ri, log_A_real,
                                                             A_imag, dtacm);
  k_film      <<<(BB * 2 * HH + 255) / 256, 256, 0, stream>>>(cond, film_w, film_b, gbuf);
  k_s4kernel  <<<HH * (LL / 256), 256, 0, stream>>>(dtacm, K_bf);

  dim3 g1(HH / 16, LL / 128, BB);
  k_gemm_gelu <<<g1, 256, 0, stream>>>(x_bf, w_bf, linear_b, h_bf);

  k_conv_epilogue<<<HH, 256, SMEM2, stream>>>(h_bf, K_bf, Dcoef, gbuf, res_w, x, out);
}
